// ProtoNet_61426622267569
// MI455X (gfx1250) — matrix operand order repaired and checked by reading, not compiled, not run on hardware
//
#include <hip/hip_runtime.h>
#include <math.h>

// ---------------------------------------------------------------------------
// CDNA5 (gfx1250, wave32) LeNet-ProtoNet pipeline, all GEMMs via
// v_wmma_f32_16x16x32_f16.  f16 operands / f32 accumulation.
// Conv layers use a window-row-padded K layout (8 slots per 5-tap row, pad
// slots carry zero WEIGHTS) so the im2col B-fragment gather is branchless:
// per fragment = 2x (v_cndmask-of-constants + add) + 2 unaligned LDS 16B
// loads (compiler merges the 8 consecutive u16 reads into ds_load_b128).
// Epilogue stores are branchless too (dump-row / padded-column sinks).
// ---------------------------------------------------------------------------

typedef _Float16 half_t;
typedef _Float16 h16 __attribute__((ext_vector_type(16)));
typedef _Float16 h8  __attribute__((ext_vector_type(8)));
typedef float    v8f __attribute__((ext_vector_type(8)));

#define CONV_TPB 128
#define FC_TPB   256
#define FC_WAVES 8

// per-window-row constant LDS offsets (c = r/5, dy = r%5); pad rows -> 0
// (pad rows read valid-but-arbitrary data; their weights are zero).
#define KOFF1(r) (((r) < 15) ? (((r) / 5) * 1024 + ((r) % 5) * 32) : 0)
#define KOFF2(r) (((r) < 30) ? (((r) / 5) * 196  + ((r) % 5) * 14) : 0)

__device__ __forceinline__ h16 hcat(h8 lo, h8 hi) {
  h16 r;
#pragma unroll
  for (int i = 0; i < 8; ++i) { r[i] = lo[i]; r[i + 8] = hi[i]; }
  return r;
}

__device__ __forceinline__ v8f wmma16(h16 a, h16 b, v8f c) {
  return __builtin_amdgcn_wmma_f32_16x16x32_f16(
      /*neg_a=*/false, a, /*neg_b=*/false, b,
      /*c_mod=*/(short)0, c, /*reuse_a=*/false, /*reuse_b=*/false);
}

// A fragment (16x32 f16): source row-major [16][lda] halfs. lda % 8 == 0,
// rows 16B aligned.  lane holds row M=lane&15, K = kt*32 + (lane>>4)*16 +
// {0..15}  -> two 16B loads.
__device__ __forceinline__ h16 load_a(const half_t* A, int lda, int kt, int lane) {
  const int row = lane & 15, hf = lane >> 4;
  const half_t* p = A + row * lda + kt * 32 + hf * 16;
  return hcat(*(const h8*)p, *(const h8*)(p + 8));
}

// B fragment (32x16 f16): source stored as [n][k] row-major (W[out][in]).
// lane holds column N=lane&15, K = kt*32 + (lane>>4)*16 + {0..15}.
__device__ __forceinline__ h16 load_b(const half_t* B, int ldb, int kt, int lane) {
  const int n = lane & 15, g = lane >> 4;
  const half_t* p = B + n * ldb + kt * 32 + g * 16;
  return hcat(*(const h8*)p, *(const h8*)(p + 8));
}

// Branchless conv B-fragment: two 8-element runs at LDS half-offsets b0, b1.
__device__ __forceinline__ h16 gather2(const half_t* s, int b0, int b1) {
  h16 r;
#pragma unroll
  for (int e = 0; e < 8; ++e) r[e] = s[b0 + e];
#pragma unroll
  for (int e = 0; e < 8; ++e) r[8 + e] = s[b1 + e];
  return r;
}

// ---------------------------------------------------------------------------
// K0a: f32 -> f16 padding for FC weights (row-major [out][in] kept as-is).
// ---------------------------------------------------------------------------
__global__ void prep_pad_f16(const float* __restrict__ src, half_t* __restrict__ dst,
                             int ru, int cu, int rp, int cp) {
  const int tot = rp * cp;
  for (int i = blockIdx.x * blockDim.x + threadIdx.x; i < tot;
       i += gridDim.x * blockDim.x) {
    const int r = i / cp, c = i - r * cp;
    dst[i] = (r < ru && c < cu) ? (half_t)src[r * cu + c] : (half_t)0.f;
  }
}

// ---------------------------------------------------------------------------
// K0b: conv weights into window-row-padded layout:
//   dst[16][cp], k' = (c*5+dy)*8 + dx ; dx in [5,8) and padded rows are ZERO
//   (these zeros neutralize the garbage activations the gather over-reads).
// ---------------------------------------------------------------------------
__global__ void prep_conv_w(const float* __restrict__ src, half_t* __restrict__ dst,
                            int oc, int ic, int cp) {
  const int tot = 16 * cp;
  for (int i = blockIdx.x * blockDim.x + threadIdx.x; i < tot;
       i += gridDim.x * blockDim.x) {
    const int r = i / cp, kq = i - r * cp;
    const int row = kq >> 3, dx = kq & 7;
    const int c = row / 5, dy = row - 5 * c;
    half_t v = (half_t)0.f;
    if (r < oc && row < ic * 5 && dx < 5)
      v = (half_t)src[r * (ic * 25) + c * 25 + dy * 5 + dx];
    dst[i] = v;
  }
}

// ---------------------------------------------------------------------------
// K1: fused conv1(3->6)+relu+pool2 + conv2(6->16)+relu+pool2, one WG/image.
//   conv1: M=16(6) x N=784 (49 tiles) x K'=128 (15 window rows) -> 196 WMMA
//   conv2: M=16    x N=112 (7 tiles)  x K'=256 (30 window rows) ->  56 WMMA
// Output: padded flattened feature row X[img][416] (400 used), f16.
// ---------------------------------------------------------------------------
__global__ void __launch_bounds__(CONV_TPB)
lenet_conv(const float* __restrict__ x,
           const half_t* __restrict__ W1, const float* __restrict__ cb1,
           const half_t* __restrict__ W2, const float* __restrict__ cb2,
           half_t* __restrict__ X) {
  __shared__ __align__(16) half_t simg[3 * 1024 + 16];   // f16 image (+overread pad)
  __shared__ __align__(16) float  co1[7 * 784];          // conv1 out + dump row
  __shared__ __align__(16) half_t sh2[6 * 14 * 14 + 32]; // pooled (+overread pad)
  __shared__ __align__(16) float  co2[16 * 112];         // conv2 out (padded cols)

  const int tid = threadIdx.x, lane = tid & 31, wave = tid >> 5;
  const int img = blockIdx.x;
  const int m0 = (lane >> 4) * 8;
  const int g  = lane >> 4;

  // hoisted biases (one global read burst per lane)
  float b1v[8], b2v[8];
#pragma unroll
  for (int v = 0; v < 8; ++v) {
    const int ch = m0 + v;
    b1v[v] = (ch < 6) ? cb1[ch] : 0.f;
    b2v[v] = cb2[ch];
  }

  const float* xin = x + (size_t)img * 3072;
  for (int i = tid; i < 3072; i += CONV_TPB) simg[i] = (half_t)xin[i];
  for (int i = 3072 + tid; i < 3 * 1024 + 16; i += CONV_TPB) simg[i] = (half_t)0.f;
  __syncthreads();

  { // conv1
    h16 aw[4];
#pragma unroll
    for (int kt = 0; kt < 4; ++kt) aw[kt] = load_a(W1, 128, kt, lane);
    for (int t = wave; t < 49; t += CONV_TPB / 32) {
      const int pix = t * 16 + (lane & 15);          // < 784 always
      const int oy = pix / 28, ox = pix - 28 * oy;
      const int s0 = oy * 32 + ox;                   // runtime pixel base
      v8f acc = {};
#pragma unroll
      for (int kt = 0; kt < 4; ++kt) {               // constants fold per kt
        const int b0 = s0 + (g ? KOFF1(kt * 4 + 2) : KOFF1(kt * 4 + 0));
        const int b1 = s0 + (g ? KOFF1(kt * 4 + 3) : KOFF1(kt * 4 + 1));
        acc = wmma16(aw[kt], gather2(simg, b0, b1), acc);
      }
#pragma unroll
      for (int v = 0; v < 8; ++v) {
        // branchless: channels >=6 (incl. whole upper half-wave) go to dump
        // row 6; v is a literal so this is a cndmask between two constants.
        const int chp = (g || v >= 6) ? 6 : v;
        co1[chp * 784 + pix] = fmaxf(acc[v] + b1v[v], 0.f);
      }
    }
  }
  __syncthreads();
  for (int i = tid; i < 6 * 14 * 14; i += CONV_TPB) {  // 2x2 maxpool
    const int ch = i / 196, r = (i % 196) / 14, c = i % 14;
    const float* p = &co1[ch * 784 + 2 * r * 28 + 2 * c];
    sh2[i] = (half_t)fmaxf(fmaxf(p[0], p[1]), fmaxf(p[28], p[29]));
  }
  for (int i = 6 * 14 * 14 + tid; i < 6 * 14 * 14 + 32; i += CONV_TPB)
    sh2[i] = (half_t)0.f;
  __syncthreads();

  { // conv2
    h16 aw[8];
#pragma unroll
    for (int kt = 0; kt < 8; ++kt) aw[kt] = load_a(W2, 256, kt, lane);
    for (int t = wave; t < 7; t += CONV_TPB / 32) {
      const int pix = t * 16 + (lane & 15);          // < 112; cols 100..111 pad
      const bool ok = pix < 100;
      const int oy = ok ? pix / 10 : 0;
      const int ox = ok ? pix - 10 * (pix / 10) : 0;
      const int s0 = oy * 14 + ox;
      v8f acc = {};
#pragma unroll
      for (int kt = 0; kt < 8; ++kt) {
        const int b0 = s0 + (g ? KOFF2(kt * 4 + 2) : KOFF2(kt * 4 + 0));
        const int b1 = s0 + (g ? KOFF2(kt * 4 + 3) : KOFF2(kt * 4 + 1));
        acc = wmma16(aw[kt], gather2(sh2, b0, b1), acc);
      }
#pragma unroll
      for (int v = 0; v < 8; ++v)                    // unconditional: pix<112
        co2[(m0 + v) * 112 + pix] = fmaxf(acc[v] + b2v[v], 0.f);
    }
  }
  __syncthreads();
  half_t* xr = X + (size_t)img * 416;
  for (int i = tid; i < 416; i += CONV_TPB) {          // 2x2 maxpool -> row
    half_t v = (half_t)0.f;
    if (i < 400) {
      const int ch = i / 25, r = (i % 25) / 5, c = i % 5;
      const float* p = &co2[ch * 112 + 2 * r * 10 + 2 * c];
      v = (half_t)fmaxf(fmaxf(p[0], p[1]), fmaxf(p[10], p[11]));
    }
    xr[i] = v;
  }
}

// ---------------------------------------------------------------------------
// K2: fused fc1(400->120)+relu -> fc2(120->84)+relu -> fc3(84->64).
// One wave per 16-row batch tile; intermediates in per-wave LDS slices,
// weight B-fragments stream from L2.  Emits deterministic per-tile partial
// sums of the embeddings for the prototype reduction.
// ---------------------------------------------------------------------------
__global__ void __launch_bounds__(FC_TPB)
lenet_fc(const half_t* __restrict__ X,
         const half_t* __restrict__ F1, const float* __restrict__ fb1,
         const half_t* __restrict__ F2, const float* __restrict__ fb2,
         const half_t* __restrict__ F3, const float* __restrict__ fb3,
         float* __restrict__ emb, float* __restrict__ partials, int nTiles) {
  __shared__ __align__(16) half_t Y1[FC_WAVES][16 * 128];
  __shared__ __align__(16) half_t Y2[FC_WAVES][16 * 96];
  const int lane = threadIdx.x & 31, wave = threadIdx.x >> 5;
  const int rt = blockIdx.x * FC_WAVES + wave;
  if (rt >= nTiles) return;                // per-wave exit; no block barriers
  const int n = lane & 15, m0 = (lane >> 4) * 8;
  half_t* y1 = &Y1[wave][0];
  half_t* y2 = &Y2[wave][0];

  { // fc1: [16x416] x [416x128]
    const half_t* A0 = X + (size_t)rt * (16 * 416);
    h16 af[13];
#pragma unroll
    for (int kt = 0; kt < 13; ++kt) af[kt] = load_a(A0, 416, kt, lane);
#pragma unroll
    for (int nt = 0; nt < 8; ++nt) {
      v8f acc = {};
      const half_t* Bn = F1 + nt * (16 * 416);
#pragma unroll
      for (int kt = 0; kt < 13; ++kt)
        acc = wmma16(af[kt], load_b(Bn, 416, kt, lane), acc);
      const int u = nt * 16 + n;
      const float bias = (u < 120) ? fb1[u] : 0.f;
#pragma unroll
      for (int v = 0; v < 8; ++v) {
        const float val = (u < 120) ? fmaxf(acc[v] + bias, 0.f) : 0.f;
        y1[(m0 + v) * 128 + u] = (half_t)val;
      }
    }
  }
  asm volatile("s_wait_dscnt 0" ::: "memory");   // LDS RAW fence (DScnt)
  { // fc2: [16x128] x [128x96]
    h16 af[4];
#pragma unroll
    for (int kt = 0; kt < 4; ++kt) af[kt] = load_a(y1, 128, kt, lane);
#pragma unroll
    for (int nt = 0; nt < 6; ++nt) {
      v8f acc = {};
      const half_t* Bn = F2 + nt * (16 * 128);
#pragma unroll
      for (int kt = 0; kt < 4; ++kt)
        acc = wmma16(af[kt], load_b(Bn, 128, kt, lane), acc);
      const int u = nt * 16 + n;
      const float bias = (u < 84) ? fb2[u] : 0.f;
#pragma unroll
      for (int v = 0; v < 8; ++v) {
        const float val = (u < 84) ? fmaxf(acc[v] + bias, 0.f) : 0.f;
        y2[(m0 + v) * 96 + u] = (half_t)val;
      }
    }
  }
  asm volatile("s_wait_dscnt 0" ::: "memory");
  float s = 0.f;
  { // fc3: [16x96] x [96x64] -> embeddings (f32)
    h16 af[3];
#pragma unroll
    for (int kt = 0; kt < 3; ++kt) af[kt] = load_a(y2, 96, kt, lane);
    float* erow = emb + (size_t)rt * (16 * 64);
#pragma unroll
    for (int nt = 0; nt < 4; ++nt) {
      v8f acc = {};
      const half_t* Bn = F3 + nt * (16 * 96);
#pragma unroll
      for (int kt = 0; kt < 3; ++kt)
        acc = wmma16(af[kt], load_b(Bn, 96, kt, lane), acc);
      const int u = nt * 16 + n;
      const float bias = fb3[u];
#pragma unroll
      for (int v = 0; v < 8; ++v) {
        const float val = acc[v] + bias;
        erow[(m0 + v) * 64 + u] = val;
        s += val;
      }
    }
  }
#pragma unroll
  for (int off = 16; off > 0; off >>= 1) s += __shfl_xor(s, off, 32);
  if (lane == 0) partials[rt] = s;
}

// ---------------------------------------------------------------------------
// K3: deterministic fixed-order reduction of per-tile sums -> proto = sum/5.
// ---------------------------------------------------------------------------
__global__ void reduce_proto(const float* __restrict__ partials,
                             float* __restrict__ proto, int nPart) {
  __shared__ float sb[256];
  const int tid = threadIdx.x;
  float s = 0.f;
  for (int i = tid; i < nPart; i += 256) s += partials[i];
  sb[tid] = s;
  __syncthreads();
  for (int o = 128; o > 0; o >>= 1) {
    if (tid < o) sb[tid] += sb[tid + o];
    __syncthreads();
  }
  if (tid == 0) *proto = sb[0] * 0.2f;   // / NUM_CLASSES (5)
}

// ---------------------------------------------------------------------------
// K4: row softmax over 64 logits + |softmax - proto| (== sqrt((.)^2)).
// One thread per row, row fully register-resident (float4 traffic).
// ---------------------------------------------------------------------------
__global__ void softmax_absdiff(const float* __restrict__ emb,
                                const float* __restrict__ proto,
                                float* __restrict__ out, int nRows) {
  const int row = blockIdx.x * blockDim.x + threadIdx.x;
  if (row >= nRows) return;
  const float4* e = (const float4*)(emb + (size_t)row * 64);
  float4* o = (float4*)(out + (size_t)row * 64);
  const float p = *proto;
  float b[64];
#pragma unroll
  for (int i = 0; i < 16; ++i) {
    float4 q = e[i];
    b[4*i] = q.x; b[4*i+1] = q.y; b[4*i+2] = q.z; b[4*i+3] = q.w;
  }
  float mx = b[0];
#pragma unroll
  for (int i = 1; i < 64; ++i) mx = fmaxf(mx, b[i]);
  float s = 0.f;
#pragma unroll
  for (int i = 0; i < 64; ++i) { b[i] = __expf(b[i] - mx); s += b[i]; }
  const float inv = 1.f / s;
#pragma unroll
  for (int i = 0; i < 16; ++i) {
    float4 q;
    q.x = fabsf(b[4*i]   * inv - p);
    q.y = fabsf(b[4*i+1] * inv - p);
    q.z = fabsf(b[4*i+2] * inv - p);
    q.w = fabsf(b[4*i+3] * inv - p);
    o[i] = q;
  }
}

// ---------------------------------------------------------------------------
// Workspace layout (bytes):
//   0        : proto (f32)
//   256      : partials[nTiles] (f32)
//   8192     : W1c  f16 [16][128]  (window-row padded)
//   12288    : W2c  f16 [16][256]  (window-row padded)
//   20480    : F1c  f16 [128][416]
//   126976   : F2c  f16 [96][128]
//   151552   : F3c  f16 [64][96]
//   163840   : Xws  f16 [N][416]
//   + N*832  : embws f32 [N][64]
// Total ~22.5 MB for N=20480.
// ---------------------------------------------------------------------------
extern "C" void kernel_launch(void* const* d_in, const int* in_sizes, int n_in,
                              void* d_out, int out_size, void* d_ws, size_t ws_size,
                              hipStream_t stream) {
  (void)n_in; (void)out_size; (void)ws_size;
  const float* x   = (const float*)d_in[0];
  const float* c1w = (const float*)d_in[1];
  const float* c1b = (const float*)d_in[2];
  const float* c2w = (const float*)d_in[3];
  const float* c2b = (const float*)d_in[4];
  const float* f1w = (const float*)d_in[5];
  const float* f1b = (const float*)d_in[6];
  const float* f2w = (const float*)d_in[7];
  const float* f2b = (const float*)d_in[8];
  const float* f3w = (const float*)d_in[9];
  const float* f3b = (const float*)d_in[10];

  const int N = in_sizes[0] / (3 * 32 * 32);   // 20480
  const int nTiles = N / 16;                   // 1280 (N % 16 == 0)

  char* ws = (char*)d_ws;
  float*  proto    = (float*)(ws + 0);
  float*  partials = (float*)(ws + 256);
  half_t* W1c = (half_t*)(ws + 8192);
  half_t* W2c = (half_t*)(ws + 12288);
  half_t* F1c = (half_t*)(ws + 20480);
  half_t* F2c = (half_t*)(ws + 126976);
  half_t* F3c = (half_t*)(ws + 151552);
  half_t* Xws = (half_t*)(ws + 163840);
  float*  embws = (float*)(ws + 163840 + (size_t)N * 416 * 2);

  prep_conv_w<<<8,    256, 0, stream>>>(c1w, W1c,  6, 3, 128);
  prep_conv_w<<<16,   256, 0, stream>>>(c2w, W2c, 16, 6, 256);
  prep_pad_f16<<<208, 256, 0, stream>>>(f1w, F1c, 120, 400, 128, 416);
  prep_pad_f16<<<48,  256, 0, stream>>>(f2w, F2c,  84, 120,  96, 128);
  prep_pad_f16<<<24,  256, 0, stream>>>(f3w, F3c,  64,  84,  64,  96);

  lenet_conv<<<N, CONV_TPB, 0, stream>>>(x, W1c, c1b, W2c, c2b, Xws);
  lenet_fc<<<(nTiles + FC_WAVES - 1) / FC_WAVES, FC_TPB, 0, stream>>>(
      Xws, F1c, f1b, F2c, f2b, F3c, f3b, embws, partials, nTiles);
  reduce_proto<<<1, 256, 0, stream>>>(partials, proto, nTiles);
  softmax_absdiff<<<(N + 255) / 256, 256, 0, stream>>>(embws, proto,
                                                       (float*)d_out, N);
}
